// MultiHeadAttention_54202487276082
// MI455X (gfx1250) — matrix operand order repaired and checked by reading, not compiled, not run on hardware
//
#include <hip/hip_runtime.h>
#include <hip/hip_bf16.h>

typedef __bf16 bf16_t;
typedef __attribute__((ext_vector_type(16))) __bf16 v16bf;
typedef __attribute__((ext_vector_type(8)))  __bf16 v8bf;
typedef __attribute__((ext_vector_type(4)))  __bf16 v4bf;
typedef __attribute__((ext_vector_type(8)))  float  v8f;

#define EMBED 1024
#define NHEAD 16
#define HDIM  64
#define BATCH 4
#define SEQ   2048

// ---------------------------------------------------------------------------
// WMMA fragment helpers (layouts per CDNA5 ISA 7.12.2, wave32)
// ---------------------------------------------------------------------------

// A-matrix 16x32 bf16: lane L holds row m=L&15; lane group (L>>4) selects
// K-halves {kb..kb+7, 16+kb..16+kb+7} with kb = (L>>4)*8. Contiguous 16B runs.
__device__ __forceinline__ v16bf load_a_bf16(const bf16_t* __restrict__ a0,
                                             int ld, int lane) {
  const int m  = lane & 15;
  const int kb = (lane >> 4) << 3;
  const bf16_t* r = a0 + (long)m * ld;
  v16bf a;
#pragma unroll
  for (int i = 0; i < 8; ++i) a[i] = r[kb + i];
#pragma unroll
  for (int i = 0; i < 8; ++i) a[8 + i] = r[16 + kb + i];
  return a;
}

// B-matrix 32x16 bf16 from a K-major source (src[n*ldk + k]):
// lane L holds column n=L&15, K values {kh..kh+7, 16+kh..16+kh+7}, kh=(L>>4)*8.
__device__ __forceinline__ v16bf load_b_kmaj(const bf16_t* c0, int ldk,
                                             int lane) {
  const int n  = lane & 15;
  const int kh = (lane >> 4) << 3;
  const bf16_t* c = c0 + (long)n * ldk + kh;
  v16bf b;
#pragma unroll
  for (int i = 0; i < 8; ++i) b[i] = c[i];
#pragma unroll
  for (int i = 0; i < 8; ++i) b[8 + i] = c[16 + i];
  return b;
}

__device__ __forceinline__ v8f wmma_bf16(v16bf a, v16bf b, v8f c) {
  // (neg_a, A, neg_b, B, c_mod, C, reuse_a, reuse_b)
  return __builtin_amdgcn_wmma_f32_16x16x32_bf16(false, a, false, b,
                                                 (short)0, c, false, false);
}

// ---------------------------------------------------------------------------
// Async global -> LDS copy (CDNA5 TDM-adjacent path, ISA 10.7 / 15.18.3):
// per-lane 16B transfer, tracked by ASYNCcnt, no VGPR round-trip.
// VDST VGPR carries the per-lane LDS byte offset (low 32 bits of the generic
// pointer, per the LDS aperture mapping).
// ---------------------------------------------------------------------------
__device__ __forceinline__ void async_ld_b128(const void* g, void* lds) {
  unsigned l = (unsigned)(size_t)lds;
  asm volatile("global_load_async_to_lds_b128 %0, %1, off"
               :: "v"(l), "v"(g) : "memory");
}
__device__ __forceinline__ void wait_async0() {
  asm volatile("s_wait_asynccnt 0x0" ::: "memory");
}

// ---------------------------------------------------------------------------
// Kernel 1a: weight transpose + fp32->bf16 convert:  wt[n*K + k] = w[k*N + n]
// ---------------------------------------------------------------------------
__global__ void mha_wt_kernel(const float* __restrict__ w,
                              bf16_t* __restrict__ wt, int K, int N) {
  long idx = (long)blockIdx.x * 256 + threadIdx.x;
  if (idx >= (long)K * N) return;
  int n = (int)(idx / K);
  int k = (int)(idx % K);
  wt[idx] = (bf16_t)w[(long)k * N + n];
}

// ---------------------------------------------------------------------------
// Kernel 1b: activation fp32 -> bf16 (same layout), 4 elements / thread.
// ---------------------------------------------------------------------------
__global__ void mha_xcvt_kernel(const float* __restrict__ x,
                                bf16_t* __restrict__ xb, long total4) {
  long i = (long)blockIdx.x * 256 + threadIdx.x;
  if (i >= total4) return;
  const float4 v = ((const float4*)x)[i];
  v4bf o;
  o[0] = (bf16_t)v.x; o[1] = (bf16_t)v.y;
  o[2] = (bf16_t)v.z; o[3] = (bf16_t)v.w;
  ((v4bf*)xb)[i] = o;
}

// ---------------------------------------------------------------------------
// Kernel 2: QKV projection GEMM.  x_bf(8192x1024) @ wqkv_t + b_qkv,
// scattering into q_ws[b,h,s,d], k_ws[b,h,s,d], v_ws[b,h,d,s] (bf16).
// Block = 8 waves; wave tile 32x64 (2x4 WMMA); block tile 128x128.
// ---------------------------------------------------------------------------
__device__ __forceinline__ void qkv_scatter(v8f acc, int rowBase, int colBase,
                                            int lane,
                                            const float* __restrict__ bias,
                                            bf16_t* __restrict__ q_ws,
                                            bf16_t* __restrict__ k_ws,
                                            bf16_t* __restrict__ v_ws) {
  const int grp = lane >> 4, n = lane & 15;
  const int col = colBase + n;
  const float bv = bias[col];
  const int part = col >> 10;          // 0=q 1=k 2=v
  const int h    = (col >> 6) & 15;
  const int d    = col & 63;
#pragma unroll
  for (int r = 0; r < 8; ++r) {
    int row = rowBase + grp * 8 + r;   // global token index b*SEQ+s
    int b = row >> 11;
    int s = row & (SEQ - 1);
    float val = acc[r] + bv;
    long bh = (long)(b * NHEAD + h);
    if (part == 0)      q_ws[(bh * SEQ + s) * HDIM + d] = (bf16_t)val;
    else if (part == 1) k_ws[(bh * SEQ + s) * HDIM + d] = (bf16_t)val;
    else                v_ws[(bh * HDIM + d) * SEQ + s] = (bf16_t)val;
  }
}

__global__ __launch_bounds__(256) void mha_qkv_gemm_kernel(
    const bf16_t* __restrict__ xb, const bf16_t* __restrict__ wt,
    const float* __restrict__ bias, bf16_t* __restrict__ q_ws,
    bf16_t* __restrict__ k_ws, bf16_t* __restrict__ v_ws) {
  const int lane = threadIdx.x & 31;
  const int wid  = threadIdx.x >> 5;
  const int rowBase = blockIdx.y * 128 + (wid & 3) * 32;
  const int colBase = blockIdx.x * 128 + (wid >> 2) * 64;
  v8f acc[2][4] = {};
  for (int kk = 0; kk < EMBED; kk += 32) {
    v16bf a0 = load_a_bf16(xb + (long)rowBase * EMBED + kk, EMBED, lane);
    v16bf a1 = load_a_bf16(xb + (long)(rowBase + 16) * EMBED + kk, EMBED, lane);
#pragma unroll
    for (int j = 0; j < 4; ++j) {
      v16bf bj =
          load_b_kmaj(wt + (long)(colBase + j * 16) * EMBED + kk, EMBED, lane);
      acc[0][j] = wmma_bf16(a0, bj, acc[0][j]);
      acc[1][j] = wmma_bf16(a1, bj, acc[1][j]);
    }
  }
#pragma unroll
  for (int ti = 0; ti < 2; ++ti)
#pragma unroll
    for (int j = 0; j < 4; ++j)
      qkv_scatter(acc[ti][j], rowBase + ti * 16, colBase + j * 16, lane, bias,
                  q_ws, k_ws, v_ws);
}

// ---------------------------------------------------------------------------
// Kernel 3: flash attention with double-buffered ASYNC LDS staging.
// One wave per 16-row q tile; the block's 8 waves share (b,h), so K/V tiles
// are staged to LDS ONCE per 32-key step via global_load_async_to_lds_b128
// (ASYNCcnt), fully overlapped with the WMMA/softmax work:
//   issue next K(32x64)/V(64x32,d-major) tile (1 async b128/thread/tile)
//   -> 4 WMMA QK^T from LDS -> online softmax (half-group shfl reductions)
//   -> P via padded LDS (C-layout -> A-layout, s_wait_dscnt)
//   -> 4 WMMA P.V from LDS -> s_wait_asynccnt 0 -> __syncthreads.
// Bank-skewed strides (72/40 bf16) keep rows 16B-aligned and fragment reads
// conflict-free.
// ---------------------------------------------------------------------------
__global__ __launch_bounds__(256) void mha_attn_kernel(
    const bf16_t* __restrict__ q_ws, const bf16_t* __restrict__ k_ws,
    const bf16_t* __restrict__ v_ws, const int* __restrict__ mask,
    bf16_t* __restrict__ o_ws) {
  __shared__ alignas(16) bf16_t kT[2][32][72];  // [buf][key][d]
  __shared__ alignas(16) bf16_t vT[2][64][40];  // [buf][d][key]
  __shared__ alignas(16) bf16_t pT[8][16][40];  // per-wave P relayout tile
  const int t    = threadIdx.x;
  const int lane = t & 31;
  const int wid  = t >> 5;
  const int h = blockIdx.y, b = blockIdx.z;
  const int qBase = blockIdx.x * 128 + wid * 16;
  const long bh = (long)(b * NHEAD + h);
  const bf16_t* qp = q_ws + (bh * SEQ + qBase) * HDIM;
  const bf16_t* kp = k_ws + bh * SEQ * HDIM;
  const bf16_t* vp = v_ws + bh * HDIM * SEQ;
  const int*    mp = mask + (long)b * SEQ * SEQ;
  const int grp = lane >> 4, n = lane & 15;

  // staging assignments: one 16B chunk per thread per tile
  const int kRow = t >> 3, kCol = (t & 7) * 8;   // K tile: 32 x 64
  const int vRow = t >> 2, vCol = (t & 3) * 8;   // V tile: 64 x 32

  const v16bf aQ0 = load_a_bf16(qp, HDIM, lane);        // d = 0..31
  const v16bf aQ1 = load_a_bf16(qp + 32, HDIM, lane);   // d = 32..63

  v8f o0 = {}, o1 = {}, o2 = {}, o3 = {};
  float mrun[8], lrun[8];
#pragma unroll
  for (int r = 0; r < 8; ++r) { mrun[r] = -1e30f; lrun[r] = 0.f; }

  // prologue: stage first tile
  async_ld_b128(kp + (long)kRow * HDIM + kCol, &kT[0][kRow][kCol]);
  async_ld_b128(vp + (long)vRow * SEQ + vCol, &vT[0][vRow][vCol]);
  wait_async0();
  __syncthreads();

  int p = 0;
  for (int kb = 0; kb < SEQ; kb += 32) {
    // stage next tile into the other buffer (overlaps with compute below)
    if (kb + 32 < SEQ) {
      async_ld_b128(kp + (long)(kb + 32 + kRow) * HDIM + kCol,
                    &kT[p ^ 1][kRow][kCol]);
      async_ld_b128(vp + (long)vRow * SEQ + kb + 32 + vCol,
                    &vT[p ^ 1][vRow][vCol]);
    }

    v8f s0 = {}, s1 = {};
    {
      v16bf bk = load_b_kmaj(&kT[p][0][0], 72, lane);    // keys 0-15, d 0-31
      s0 = wmma_bf16(aQ0, bk, s0);
      bk = load_b_kmaj(&kT[p][0][32], 72, lane);         // keys 0-15, d 32-63
      s0 = wmma_bf16(aQ1, bk, s0);
      bk = load_b_kmaj(&kT[p][16][0], 72, lane);         // keys 16-31
      s1 = wmma_bf16(aQ0, bk, s1);
      bk = load_b_kmaj(&kT[p][16][32], 72, lane);
      s1 = wmma_bf16(aQ1, bk, s1);
    }
    // scale 1/sqrt(64) and apply mask
#pragma unroll
    for (int r = 0; r < 8; ++r) {
      int q = qBase + grp * 8 + r;
      int m0 = mp[(long)q * SEQ + kb + n];
      int m1 = mp[(long)q * SEQ + kb + 16 + n];
      s0[r] = m0 ? s0[r] * 0.125f : -1e30f;
      s1[r] = m1 ? s1[r] * 0.125f : -1e30f;
    }
    // online softmax: row r of the C tile lives in one 16-lane half-group
    float p0[8], p1[8];
#pragma unroll
    for (int r = 0; r < 8; ++r) {
      float mx = fmaxf(s0[r], s1[r]);
      mx = fmaxf(mx, __shfl_xor(mx, 1, 32));
      mx = fmaxf(mx, __shfl_xor(mx, 2, 32));
      mx = fmaxf(mx, __shfl_xor(mx, 4, 32));
      mx = fmaxf(mx, __shfl_xor(mx, 8, 32));
      float nm   = fmaxf(mrun[r], mx);
      float corr = __expf(mrun[r] - nm);
      p0[r] = __expf(s0[r] - nm);
      p1[r] = __expf(s1[r] - nm);
      float rs = p0[r] + p1[r];
      rs += __shfl_xor(rs, 1, 32);
      rs += __shfl_xor(rs, 2, 32);
      rs += __shfl_xor(rs, 4, 32);
      rs += __shfl_xor(rs, 8, 32);
      lrun[r] = lrun[r] * corr + rs;
      mrun[r] = nm;
      o0[r] *= corr; o1[r] *= corr; o2[r] *= corr; o3[r] *= corr;
    }
    // C-layout -> A-layout for P through per-wave LDS tile (16 x 32 bf16)
#pragma unroll
    for (int r = 0; r < 8; ++r) {
      pT[wid][grp * 8 + r][n]      = (bf16_t)p0[r];
      pT[wid][grp * 8 + r][16 + n] = (bf16_t)p1[r];
    }
    // LDS ops of a wave complete in order; fence the compiler + HW explicitly
    asm volatile("s_wait_dscnt 0x0" ::: "memory");
    v16bf aP = load_a_bf16(&pT[wid][0][0], 40, lane);
    {
      v16bf bv = load_b_kmaj(&vT[p][0][0], 40, lane);    // d 0..15
      o0 = wmma_bf16(aP, bv, o0);
      bv = load_b_kmaj(&vT[p][16][0], 40, lane);         // d 16..31
      o1 = wmma_bf16(aP, bv, o1);
      bv = load_b_kmaj(&vT[p][32][0], 40, lane);         // d 32..47
      o2 = wmma_bf16(aP, bv, o2);
      bv = load_b_kmaj(&vT[p][48][0], 40, lane);         // d 48..63
      o3 = wmma_bf16(aP, bv, o3);
    }
    wait_async0();    // our async stores to buf p^1 have landed in LDS
    __syncthreads();  // all waves done reading buf p / writing buf p^1
    p ^= 1;
  }
  // normalize and store as [b, s, h, d] (== row-major 8192 x 1024)
#pragma unroll
  for (int r = 0; r < 8; ++r) {
    int q = qBase + grp * 8 + r;
    float inv = lrun[r] > 0.f ? 1.0f / lrun[r] : 0.f;
    long base = ((long)(b * SEQ + q) * NHEAD + h) * HDIM + n;
    o_ws[base]      = (bf16_t)(o0[r] * inv);
    o_ws[base + 16] = (bf16_t)(o1[r] * inv);
    o_ws[base + 32] = (bf16_t)(o2[r] * inv);
    o_ws[base + 48] = (bf16_t)(o3[r] * inv);
  }
}

// ---------------------------------------------------------------------------
// Kernel 4: output projection GEMM. o_ws(8192x1024 bf16) @ wout_t + b_out
// -> d_out fp32. Same 32x64 wave tile as the QKV GEMM.
// ---------------------------------------------------------------------------
__device__ __forceinline__ void out_store(v8f acc, int rowBase, int colBase,
                                          int lane,
                                          const float* __restrict__ bias,
                                          float* __restrict__ out) {
  const int grp = lane >> 4, n = lane & 15;
  const int col = colBase + n;
  const float bv = bias[col];
#pragma unroll
  for (int r = 0; r < 8; ++r) {
    int row = rowBase + grp * 8 + r;
    out[(long)row * EMBED + col] = acc[r] + bv;
  }
}

__global__ __launch_bounds__(256) void mha_out_gemm_kernel(
    const bf16_t* __restrict__ a, const bf16_t* __restrict__ wt,
    const float* __restrict__ bias, float* __restrict__ out) {
  const int lane = threadIdx.x & 31;
  const int wid  = threadIdx.x >> 5;
  const int rowBase = blockIdx.y * 128 + (wid & 3) * 32;
  const int colBase = blockIdx.x * 128 + (wid >> 2) * 64;
  v8f acc[2][4] = {};
  for (int kk = 0; kk < EMBED; kk += 32) {
    v16bf a0 = load_a_bf16(a + (long)rowBase * EMBED + kk, EMBED, lane);
    v16bf a1 = load_a_bf16(a + (long)(rowBase + 16) * EMBED + kk, EMBED, lane);
#pragma unroll
    for (int j = 0; j < 4; ++j) {
      v16bf bj =
          load_b_kmaj(wt + (long)(colBase + j * 16) * EMBED + kk, EMBED, lane);
      acc[0][j] = wmma_bf16(a0, bj, acc[0][j]);
      acc[1][j] = wmma_bf16(a1, bj, acc[1][j]);
    }
  }
#pragma unroll
  for (int ti = 0; ti < 2; ++ti)
#pragma unroll
    for (int j = 0; j < 4; ++j)
      out_store(acc[ti][j], rowBase + ti * 16, colBase + j * 16, lane, bias,
                out);
}

// ---------------------------------------------------------------------------
// Host launcher
// ---------------------------------------------------------------------------
extern "C" void kernel_launch(void* const* d_in, const int* in_sizes, int n_in,
                              void* d_out, int out_size, void* d_ws,
                              size_t ws_size, hipStream_t stream) {
  (void)in_sizes; (void)n_in; (void)out_size; (void)ws_size;
  const float* x      = (const float*)d_in[0];
  const int*   mask   = (const int*)d_in[1];
  const float* w_qkv  = (const float*)d_in[2];
  const float* b_qkv  = (const float*)d_in[3];
  const float* w_out  = (const float*)d_in[4];
  const float* b_out  = (const float*)d_in[5];
  float* out = (float*)d_out;

  char* ws = (char*)d_ws;
  const size_t MB = (size_t)1 << 20;
  bf16_t* q_ws   = (bf16_t*)(ws + 0 * MB);   // 16 MB  [b,h,s,d]
  bf16_t* k_ws   = (bf16_t*)(ws + 16 * MB);  // 16 MB  [b,h,s,d]
  bf16_t* v_ws   = (bf16_t*)(ws + 32 * MB);  // 16 MB  [b,h,d,s]
  bf16_t* o_ws   = (bf16_t*)(ws + 48 * MB);  // 16 MB  [b,s,h,d]
  bf16_t* wqkv_t = (bf16_t*)(ws + 64 * MB);  // 6 MB   [n,k]
  bf16_t* wout_t = (bf16_t*)(ws + 70 * MB);  // 2 MB   [n,k]
  bf16_t* x_bf   = (bf16_t*)(ws + 72 * MB);  // 16 MB  [row, k]

  {
    long tot = (long)EMBED * 3 * EMBED;
    mha_wt_kernel<<<(int)((tot + 255) / 256), 256, 0, stream>>>(
        w_qkv, wqkv_t, EMBED, 3 * EMBED);
  }
  {
    long tot = (long)EMBED * EMBED;
    mha_wt_kernel<<<(int)((tot + 255) / 256), 256, 0, stream>>>(
        w_out, wout_t, EMBED, EMBED);
  }
  {
    long tot4 = (long)BATCH * SEQ * EMBED / 4;
    mha_xcvt_kernel<<<(int)((tot4 + 255) / 256), 256, 0, stream>>>(x, x_bf,
                                                                   tot4);
  }
  mha_qkv_gemm_kernel<<<dim3(3 * EMBED / 128, BATCH * SEQ / 128), 256, 0,
                        stream>>>(x_bf, wqkv_t, b_qkv, q_ws, k_ws, v_ws);
  mha_attn_kernel<<<dim3(SEQ / 128, NHEAD, BATCH), 256, 0, stream>>>(
      q_ws, k_ws, v_ws, mask, o_ws);
  mha_out_gemm_kernel<<<dim3(EMBED / 128, BATCH * SEQ / 128), 256, 0, stream>>>(
      o_ws, wout_t, b_out, out);
}
